// DeformConv1D_50148038148681
// MI455X (gfx1250) — matrix operand order repaired and checked by reading, not compiled, not run on hardware
//
#include <hip/hip_runtime.h>
#include <stdint.h>

// ---- problem constants (match reference) ----
#define BDIM   16
#define CIN    256
#define COUT   256
#define TLEN   4096
#define KW     3
#define KDIM   (CIN * KW)      // 768
#define CHUNK  32
#define NCHUNK (KDIM / CHUNK)  // 24
#define NTILE  64              // t-values per workgroup

typedef __attribute__((ext_vector_type(16))) _Float16 v16h;
typedef __attribute__((ext_vector_type(8)))  float    v8f;

union FragU { v16h v; uint4 q[2]; };

__global__ __launch_bounds__(256)
void deform_conv1d_wmma(const float* __restrict__ x,      // (B, CIN, T)
                        const float* __restrict__ off,    // (B, 2K, T, 1)
                        const float* __restrict__ w,      // (COUT, CIN, K)
                        const float* __restrict__ bias,   // (COUT,)
                        float* __restrict__ out)          // (B, COUT, T)
{
  __shared__ __align__(16) _Float16 As[COUT * CHUNK];    // 16 KB  [m][kl]
  __shared__ __align__(16) _Float16 Bs[NTILE * CHUNK];   //  4 KB  [n][kl]
  __shared__ int   i0_s[KW * NTILE];
  __shared__ float fr_s[KW * NTILE];
  __shared__ float wx_s[KW * NTILE];

  const int tid  = threadIdx.x;
  const int wid  = tid >> 5;       // wave 0..7
  const int lane = tid & 31;
  const int l16  = lane & 15;
  const int half = lane >> 4;

  const int b     = blockIdx.x >> 6;          // 64 t-tiles per batch
  const int tbase = (blockIdx.x & 63) * NTILE;

  // ---- per-(k,t) interpolation tables: py = t + dy (k-terms cancel) ----
  if (tid < KW * NTILE) {
    const int kk = tid >> 6;      // 0..2
    const int tl = tid & 63;
    const int t  = tbase + tl;
    const float dy  = off[(b * 2 * KW + 2 * kk    ) * TLEN + t];
    const float dxv = off[(b * 2 * KW + 2 * kk + 1) * TLEN + t];
    const float py  = (float)t + dy;
    const float fl  = floorf(py);
    i0_s[tid] = (int)fl;
    fr_s[tid] = py - fl;
    wx_s[tid] = fmaxf(0.f, 1.f - fabsf(dxv));
  }
  __syncthreads();

  v8f acc[8] = {};   // [mt(2)][nt(4)] -> acc[mt*4+nt], 64 VGPRs

  const float* xb = x + (size_t)b * CIN * TLEN;
  const int pairIdx = tid & 15;    // which f16 pair of the 32-wide chunk
  const int nb      = tid >> 4;    // base n (0..15)

  for (int ch = 0; ch < NCHUNK; ++ch) {
    const int ckbase = ch * CHUNK;

    // ---- stage A: weight chunk (256 x 32) -> f16 LDS, row m = tid ----
    {
      const float* wrow = w + (size_t)tid * KDIM + ckbase;
      _Float16* arow = As + tid * CHUNK;
      #pragma unroll
      for (int j = 0; j < 8; ++j) {
        const float4 w4 = ((const float4*)wrow)[j];
        union { _Float16 h[4]; uint2 u; } pk;
        pk.h[0] = (_Float16)w4.x; pk.h[1] = (_Float16)w4.y;
        pk.h[2] = (_Float16)w4.z; pk.h[3] = (_Float16)w4.w;
        *((uint2*)arow + j) = pk.u;
      }
    }

    // ---- stage B: deformable gather (32 x 64) -> f16 LDS [n][kl] ----
    #pragma unroll
    for (int i = 0; i < 4; ++i) {
      const int n = nb + 16 * i;
      union { _Float16 h[2]; uint32_t u; } pk;
      #pragma unroll
      for (int e = 0; e < 2; ++e) {
        const int ck = ckbase + 2 * pairIdx + e;
        const int c  = ck / 3;
        const int kk = ck - 3 * c;
        const int i0 = i0_s[kk * NTILE + n];
        const float fr = fr_s[kk * NTILE + n];
        const float wx = wx_s[kk * NTILE + n];
        const float* xr = xb + (size_t)c * TLEN;
        const float x0 = ((unsigned)i0       < (unsigned)TLEN) ? xr[i0]     : 0.f;
        const float x1 = ((unsigned)(i0 + 1) < (unsigned)TLEN) ? xr[i0 + 1] : 0.f;
        pk.h[e] = (_Float16)(wx * fmaf(fr, x1 - x0, x0));
      }
      *(uint32_t*)(Bs + n * CHUNK + 2 * pairIdx) = pk.u;
    }
    __syncthreads();

    // ---- fragments + WMMA: wave wid owns m-tiles {2w, 2w+1}, all 4 n-tiles ----
    FragU a0, a1, bf;
    {
      // A layout (ISA 7.12.2, 16-bit A 16x32): lane half selects K 0-7/16-23 vs 8-15/24-31
      const char* r0 = (const char*)(As + ((wid * 2 + 0) * 16 + l16) * CHUNK);
      a0.q[0] = *(const uint4*)(r0 + half * 16);
      a0.q[1] = *(const uint4*)(r0 + half * 16 + 32);
      const char* r1 = (const char*)(As + ((wid * 2 + 1) * 16 + l16) * CHUNK);
      a1.q[0] = *(const uint4*)(r1 + half * 16);
      a1.q[1] = *(const uint4*)(r1 + half * 16 + 32);
    }
    #pragma unroll
    for (int nt = 0; nt < 4; ++nt) {
      // B layout (same K order as A): lane half selects K 0-7/16-23 vs 8-15/24-31
      const char* rb = (const char*)(Bs + (nt * 16 + l16) * CHUNK);
      bf.q[0] = *(const uint4*)(rb + half * 16);
      bf.q[1] = *(const uint4*)(rb + half * 16 + 32);
      acc[nt]     = __builtin_amdgcn_wmma_f32_16x16x32_f16(
                        false, a0.v, false, bf.v, (short)0, acc[nt],     false, false);
      acc[4 + nt] = __builtin_amdgcn_wmma_f32_16x16x32_f16(
                        false, a1.v, false, bf.v, (short)0, acc[4 + nt], false, false);
    }
    __syncthreads();
  }

  // ---- epilogue: bias + store (C/D layout: VGPR r -> M = 8*half + r, N = l16) ----
  #pragma unroll
  for (int mt = 0; mt < 2; ++mt) {
    const int mbase = (wid * 2 + mt) * 16;
    #pragma unroll
    for (int nt = 0; nt < 4; ++nt) {
      const int t = tbase + nt * 16 + l16;
      #pragma unroll
      for (int r = 0; r < 8; ++r) {
        const int o = mbase + half * 8 + r;
        out[((size_t)b * COUT + o) * TLEN + t] = acc[mt * 4 + nt][r] + bias[o];
      }
    }
  }
}

extern "C" void kernel_launch(void* const* d_in, const int* in_sizes, int n_in,
                              void* d_out, int out_size, void* d_ws, size_t ws_size,
                              hipStream_t stream) {
  (void)in_sizes; (void)n_in; (void)out_size; (void)d_ws; (void)ws_size;
  const float* x    = (const float*)d_in[0];
  const float* off  = (const float*)d_in[1];
  const float* w    = (const float*)d_in[2];
  const float* bias = (const float*)d_in[3];
  float* out = (float*)d_out;

  dim3 grid(BDIM * (TLEN / NTILE));   // 16 * 64 = 1024 workgroups
  dim3 block(256);                    // 8 wave32 waves
  deform_conv1d_wmma<<<grid, block, 0, stream>>>(x, off, w, bias, out);
}
